// SpikingEmbed_48507360641263
// MI455X (gfx1250) — matrix operand order repaired and checked by reading, not compiled, not run on hardware
//
#include <hip/hip_runtime.h>
#include <hip/hip_bf16.h>

typedef __attribute__((ext_vector_type(16))) _Float16 v16h;
typedef __attribute__((ext_vector_type(8)))  float    v8f;
typedef __attribute__((ext_vector_type(4)))  unsigned int u32x4;

#define B_   32
#define T_   2048
#define CIN  80
#define H_   256
#define KREAL 240          // real K = 3*CIN; padded to 256
#define M_   (B_ * T_)     // 65536
#define TPAD (T_ + 3)      // 1 front zero row + 2 back zero rows per batch
#define XH_STRIDE_B (TPAD * CIN)   // halfs per batch in padded fp16 x

// ---------------------------------------------------------------------------
// 1) x (B,T,Cin) fp32 -> xh (B, T+3, Cin) fp16, zero-padded rows tp=0, T+1, T+2
// ---------------------------------------------------------------------------
__global__ void pack_x_kernel(const float* __restrict__ x, _Float16* __restrict__ xh) {
    int idx = blockIdx.x * blockDim.x + threadIdx.x;
    const int n = B_ * TPAD * CIN;
    if (idx >= n) return;
    int b   = idx / (TPAD * CIN);
    int r   = idx % (TPAD * CIN);
    int tp  = r / CIN;
    int cin = r % CIN;
    float v = 0.0f;
    if (tp >= 1 && tp <= T_) v = x[(b * T_ + (tp - 1)) * CIN + cin];
    xh[idx] = (_Float16)v;
}

// ---------------------------------------------------------------------------
// 2) conv_w (H, Cin, 3) fp32 -> WMMA B-fragment layout fp16.
//    Layout: [ntile(16)][chunk(8)][lane(32)][i(16 halfs, contiguous)]
//    B(32x16, f16): lanes 0-15 -> n=lane, K={0..7,16..23}; lanes 16-31 -> n=lane-16,
//    K={8..15,24..31}; K packed in pairs per VGPR => local k = (lane>>4)*8 + (i&7) + (i>>3)*16.
//    Flattened K index j = kk*Cin + cin, matching the contiguous-window view
//    of x; j >= 240 is zero padding.
// ---------------------------------------------------------------------------
__global__ void pack_w_kernel(const float* __restrict__ w, _Float16* __restrict__ wp) {
    int idx = blockIdx.x * blockDim.x + threadIdx.x;   // 16*8*32*16 = 65536
    if (idx >= 16 * 8 * 32 * 16) return;
    int i     =  idx        & 15;
    int lane  = (idx >> 4)  & 31;
    int chunk = (idx >> 9)  & 7;
    int ntile =  idx >> 12;
    int n = ntile * 16 + (lane & 15);
    int k = chunk * 32 + ((lane >> 4) * 8) + (i & 7) + ((i >> 3) * 16);
    float v = 0.0f;
    if (k < KREAL) {
        int cin = k % CIN;
        int kk  = k / CIN;
        v = w[n * (CIN * 3) + cin * 3 + kk];
    }
    wp[idx] = (_Float16)v;
}

// ---------------------------------------------------------------------------
// 3) Implicit-GEMM conv via WMMA.
//    Block = 128 threads (4 waves) covering a 64-row M slab (4 M-tiles).
//    Each wave computes a 64x64 output tile: 4 M-tiles x 4 N-tiles,
//    16 accumulators, 16 WMMAs per K-chunk against 16 fragment loads.
// ---------------------------------------------------------------------------
__global__ void __launch_bounds__(128)
conv_gemm_kernel(const _Float16* __restrict__ xh,
                 const _Float16* __restrict__ wp,
                 float* __restrict__ y) {
    const int lane   = threadIdx.x & 31;
    const int wave   = threadIdx.x >> 5;          // 0..3  -> N range
    const int laneL  = lane & 15;
    const int laneHi = lane >> 4;                 // 0 or 1
    const int mbase  = blockIdx.x * 4;            // first M-tile of slab

    // A-matrix row pointers for the 4 M-tiles (ISA 16-bit A 16x32 layout:
    // lane L holds row m=L; K groups {0..7,16..23} lanes 0-15,
    // {8..15,24..31} lanes 16-31).
    const _Float16* rows[4];
    #pragma unroll
    for (int mt = 0; mt < 4; ++mt) {
        const int m = (mbase + mt) * 16 + laneL;
        const int b = m >> 11;                    // / T_
        const int t = m & (T_ - 1);
        rows[mt] = xh + (size_t)b * XH_STRIDE_B + (size_t)t * CIN;
    }

    v8f acc[4][4];
    #pragma unroll
    for (int mt = 0; mt < 4; ++mt)
        #pragma unroll
        for (int nt = 0; nt < 4; ++nt)
            acc[mt][nt] = v8f{0};

    #pragma unroll
    for (int chunk = 0; chunk < 8; ++chunk) {
        const int k0 = chunk * 32;
        const int ka = k0 + laneHi * 8;

        // ---- 8 A-fragment b128 loads (4 M-tiles) ----
        v16h afrag[4];
        #pragma unroll
        for (int mt = 0; mt < 4; ++mt) {
            u32x4 lo = *(const u32x4*)(rows[mt] + ka);
            u32x4 hi;
            if (k0 < 224) {                        // hi K group < 240: valid
                hi = *(const u32x4*)(rows[mt] + ka + 16);
            } else {                               // K in [240,256) -> zeros
                hi = u32x4{0u, 0u, 0u, 0u};
            }
            reinterpret_cast<u32x4*>(&afrag[mt])[0] = lo;
            reinterpret_cast<u32x4*>(&afrag[mt])[1] = hi;
        }

        // ---- 8 B-fragment b128 loads (4 N-tiles) ----
        v16h bfrag[4];
        #pragma unroll
        for (int nt = 0; nt < 4; ++nt) {
            const _Float16* bp =
                wp + (((size_t)((wave * 4 + nt) * 8 + chunk) * 32 + lane) * 16);
            reinterpret_cast<u32x4*>(&bfrag[nt])[0] = *(const u32x4*)(bp);
            reinterpret_cast<u32x4*>(&bfrag[nt])[1] = *(const u32x4*)(bp + 8);
        }

        // ---- 16 WMMAs ----
        #pragma unroll
        for (int nt = 0; nt < 4; ++nt)
            #pragma unroll
            for (int mt = 0; mt < 4; ++mt)
                acc[mt][nt] = __builtin_amdgcn_wmma_f32_16x16x32_f16(
                    false, afrag[mt], false, bfrag[nt], (short)0,
                    acc[mt][nt], false, false);
    }

    // C/D layout: lanes 0-15 -> N=lane, VGPR r -> M=r ; lanes 16-31 -> M=r+8.
    #pragma unroll
    for (int mt = 0; mt < 4; ++mt) {
        #pragma unroll
        for (int nt = 0; nt < 4; ++nt) {
            const int h = (wave * 4 + nt) * 16 + laneL;
            #pragma unroll
            for (int r = 0; r < 8; ++r) {
                const int mg = (mbase + mt) * 16 + laneHi * 8 + r;
                y[(size_t)mg * H_ + h] = acc[mt][nt][r];
            }
        }
    }
}

// ---------------------------------------------------------------------------
// 4) BN statistics: per-channel sum / sumsq over M = B*T values.
// ---------------------------------------------------------------------------
__global__ void stats_init_kernel(float* __restrict__ stats) {
    int i = blockIdx.x * blockDim.x + threadIdx.x;
    if (i < 2 * H_) stats[i] = 0.0f;
}

__global__ void bn_stats_kernel(const float* __restrict__ y, float* __restrict__ stats) {
    const int h = threadIdx.x;                 // 0..255
    const int mbase = blockIdx.x * 256;        // 256 blocks cover M
    float s = 0.0f, s2 = 0.0f;
    #pragma unroll 4
    for (int i = 0; i < 256; ++i) {
        float v = y[(size_t)(mbase + i) * H_ + h];
        s  += v;
        s2 += v * v;
    }
    atomicAdd(&stats[h], s);
    atomicAdd(&stats[H_ + h], s2);
}

// ---------------------------------------------------------------------------
// 5) Fused BN-affine + LIF scan. One thread per (b, h) sequence.
//    decay = 1 - 1/tau = 0.5 ; hard reset to 0 ; out layout (T, B, H).
// ---------------------------------------------------------------------------
__global__ void __launch_bounds__(256)
lif_kernel(const float* __restrict__ y, const float* __restrict__ stats,
           const float* __restrict__ gamma, const float* __restrict__ beta,
           float* __restrict__ out) {
    const int h = threadIdx.x;
    const int b = blockIdx.x;
    const float inv  = 1.0f / (float)M_;
    const float mean = stats[h] * inv;
    const float var  = stats[H_ + h] * inv - mean * mean;
    const float sc   = rsqrtf(var + 1e-5f) * gamma[h];
    const float sh   = beta[h] - mean * sc;

    const float* yb = y + (size_t)b * T_ * H_ + h;
    float* ob       = out + (size_t)b * H_ + h;

    float v = 0.0f;
    for (int t = 0; t < T_; ++t) {
        const float yn = yb[(size_t)t * H_] * sc + sh;
        v = v * 0.5f + yn;
        const float s = (v - 1.0f >= 0.0f) ? 1.0f : 0.0f;
        __builtin_nontemporal_store(s, ob + (size_t)t * B_ * H_);
        v = v * (1.0f - s);
    }
}

// ---------------------------------------------------------------------------
extern "C" void kernel_launch(void* const* d_in, const int* in_sizes, int n_in,
                              void* d_out, int out_size, void* d_ws, size_t ws_size,
                              hipStream_t stream) {
    const float* x      = (const float*)d_in[0];   // (B, T, Cin)
    const float* conv_w = (const float*)d_in[1];   // (H, Cin, 3)
    // conv_b (d_in[2]) is absorbed by BatchNorm's mean subtraction -> unused.
    const float* gamma  = (const float*)d_in[3];
    const float* beta   = (const float*)d_in[4];
    float* out = (float*)d_out;                    // (T, B, H) fp32 spikes

    // Workspace layout
    char* ws = (char*)d_ws;
    const size_t Y_BYTES  = (size_t)M_ * H_ * sizeof(float);             // 67.1 MB
    const size_t XH_BYTES = (size_t)B_ * XH_STRIDE_B * sizeof(_Float16); // 10.5 MB
    const size_t WP_BYTES = (size_t)16 * 8 * 32 * 16 * sizeof(_Float16); // 128 KB
    float*    y     = (float*)ws;
    _Float16* xh    = (_Float16*)(ws + Y_BYTES);
    _Float16* wp    = (_Float16*)(ws + Y_BYTES + XH_BYTES);
    float*    stats = (float*)(ws + Y_BYTES + XH_BYTES + WP_BYTES);

    // 1) pack x -> fp16 padded
    {
        const int n = B_ * TPAD * CIN;
        pack_x_kernel<<<(n + 255) / 256, 256, 0, stream>>>(x, xh);
    }
    // 2) pack weights -> WMMA B fragments
    pack_w_kernel<<<(16 * 8 * 32 * 16) / 256, 256, 0, stream>>>(conv_w, wp);
    // 3) implicit-GEMM conv via WMMA (64-row slab per block, 64x64 per wave)
    conv_gemm_kernel<<<M_ / 64, 128, 0, stream>>>(xh, wp, y);
    // 4) BN stats
    stats_init_kernel<<<2, 256, 0, stream>>>(stats);
    bn_stats_kernel<<<M_ / 256, 256, 0, stream>>>(y, stats);
    // 5) fused BN + LIF scan -> spikes
    lif_kernel<<<B_, 256, 0, stream>>>(y, stats, gamma, beta, out);
}
